// DLRM_31920196944515
// MI455X (gfx1250) — matrix operand order repaired and checked by reading, not compiled, not run on hardware
//
#include <hip/hip_runtime.h>
#include <math.h>

#define NTAB   26
#define NROWS  100000
#define EMB    64
#define BATCH  16384

typedef __attribute__((ext_vector_type(16))) _Float16 v16h;
typedef __attribute__((ext_vector_type(8)))  _Float16 v8h;   // 16 bytes
typedef __attribute__((ext_vector_type(8)))  float    v8f;

// ---------------------------------------------------------------------------
// f32 -> f16 convert with K padding (pad columns zeroed).
// src: [rows, srcK] f32 row-major   dst: [rows, dstK] f16 row-major
// ---------------------------------------------------------------------------
__global__ __launch_bounds__(256) void cvt_pad_f16(
    const float* __restrict__ src, _Float16* __restrict__ dst,
    int rows, int srcK, int dstK)
{
    long idx = (long)blockIdx.x * blockDim.x + threadIdx.x;
    long total = (long)rows * dstK;
    if (idx >= total) return;
    int r = (int)(idx / dstK);
    int k = (int)(idx - (long)r * dstK);
    dst[idx] = (k < srcK) ? (_Float16)src[(long)r * srcK + k] : (_Float16)0.f;
}

// ---------------------------------------------------------------------------
// Generic GEMM: out[m,n] = act( A[m,:] . Wt[n,:] + bias[n] )
// A:  [M, lda] f16 (K contiguous)          Wt: [N, ldw] f16 (K contiguous)
// out:[M, ldo] f16.
// Each wave computes a 32x64 block = 2 M-tiles x 4 N-tiles of 16x16:
// A fragments reused across 4 N-tiles, B fragments across 2 M-tiles
// (~21 FLOP/byte of L2 traffic vs 8 for the 1-tile version).
// grid = (N/64, M/256), block = 256 (8 waves).
// ---------------------------------------------------------------------------
__global__ __launch_bounds__(256) void gemm_wmma_relu(
    const _Float16* __restrict__ A, int lda,
    const _Float16* __restrict__ Wt, int ldw,
    const float* __restrict__ bias,
    _Float16* __restrict__ out, int ldo,
    int K, int relu)
{
    const int lane = threadIdx.x & 31;
    const int wave = threadIdx.x >> 5;
    const int tn0  = blockIdx.x * 4;             // first of 4 N tiles
    const int tm0  = blockIdx.y * 16 + wave * 2; // first of 2 M tiles
    const int l15  = lane & 15;

    // A fragment (ISA 16-bit A 16x32 layout): lane L = row L%16;
    //  halves 0-7 = K[koff..koff+7], halves 8-15 = K[16+koff..]; koff = (L>=16)?8:0
    const int akoff = (lane >= 16) ? 8 : 0;
    // B fragment (ISA 16-bit B 32x16 layout): lane L = col L%16;
    //  halves 0-7 = K[akoff..akoff+7], halves 8-15 = K[16+akoff..] (as A); bkoff unused
    const int bkoff = (lane >= 16) ? 16 : 0;

    const _Float16* arow0 = A + (size_t)(tm0 * 16 + l15) * lda;
    const _Float16* arow1 = arow0 + (size_t)16 * lda;
    const _Float16* brow[4];
#pragma unroll
    for (int j = 0; j < 4; ++j)
        brow[j] = Wt + (size_t)((tn0 + j) * 16 + l15) * ldw;

    v8f acc[2][4];
#pragma unroll
    for (int mi = 0; mi < 2; ++mi)
#pragma unroll
        for (int j = 0; j < 4; ++j)
            acc[mi][j] = (v8f){0.f, 0.f, 0.f, 0.f, 0.f, 0.f, 0.f, 0.f};

    for (int k0 = 0; k0 < K; k0 += 32) {
        union { v16h v; v8h h[2]; } a0, a1, b;
        a0.h[0] = *(const v8h*)(arow0 + k0 + akoff);
        a0.h[1] = *(const v8h*)(arow0 + k0 + 16 + akoff);
        a1.h[0] = *(const v8h*)(arow1 + k0 + akoff);
        a1.h[1] = *(const v8h*)(arow1 + k0 + 16 + akoff);
#pragma unroll
        for (int j = 0; j < 4; ++j) {
            b.h[0] = *(const v8h*)(brow[j] + k0 + akoff);
            b.h[1] = *(const v8h*)(brow[j] + k0 + 16 + akoff);
            acc[0][j] = __builtin_amdgcn_wmma_f32_16x16x32_f16(
                            false, a0.v, false, b.v, (short)0, acc[0][j], false, false);
            acc[1][j] = __builtin_amdgcn_wmma_f32_16x16x32_f16(
                            false, a1.v, false, b.v, (short)0, acc[1][j], false, false);
        }
    }

    // C layout: VGPR r -> row tile*16 + r (+8 for lanes 16-31), col = tile_n*16 + lane%16
#pragma unroll
    for (int j = 0; j < 4; ++j) {
        const int col = (tn0 + j) * 16 + l15;
        const float bv = bias[col];
#pragma unroll
        for (int mi = 0; mi < 2; ++mi) {
            const int rbase = (tm0 + mi) * 16 + ((lane >= 16) ? 8 : 0);
#pragma unroll
            for (int r = 0; r < 8; ++r) {
                float v = acc[mi][j][r] + bv;
                if (relu) v = fmaxf(v, 0.f);
                out[(size_t)(rbase + r) * ldo + col] = (_Float16)v;
            }
        }
    }
}

// ---------------------------------------------------------------------------
// Embedding bags: one wave per 2 consecutive bags (same table: BATCH even).
// Prefetch the second bag's rows (global_prefetch_b8) before consuming the
// first so consecutive random gathers overlap. Sum 2 rows of 64 f32 -> f16,
// write into Tbuf[b][1+t][:]  (Tbuf is [B, 32, 64] f16)
// ---------------------------------------------------------------------------
__global__ __launch_bounds__(256) void embed_bag(
    const float* __restrict__ E, const int* __restrict__ lS_i,
    _Float16* __restrict__ Tbuf)
{
    const int lane = threadIdx.x & 31;
    const int wid  = (int)((blockIdx.x * (unsigned)blockDim.x + threadIdx.x) >> 5);
    const int bag0 = wid * 2;
    const int t = bag0 / BATCH;          // both bags in same table
    const int b0 = bag0 - t * BATCH;

    const int* idx = lS_i + (size_t)t * (BATCH * 2) + 2 * b0;
    const int iA0 = idx[0], iA1 = idx[1], iB0 = idx[2], iB1 = idx[3];

    const float* pA0 = E + ((size_t)t * NROWS + iA0) * EMB + lane * 2;
    const float* pA1 = E + ((size_t)t * NROWS + iA1) * EMB + lane * 2;
    const float* pB0 = E + ((size_t)t * NROWS + iB0) * EMB + lane * 2;
    const float* pB1 = E + ((size_t)t * NROWS + iB1) * EMB + lane * 2;

    __builtin_prefetch(pB0, 0, 0);
    __builtin_prefetch(pB1, 0, 0);

    union { unsigned int u; _Float16 h[2]; } pk;
    {
        const float2 a = *(const float2*)pA0;
        const float2 c = *(const float2*)pA1;
        pk.h[0] = (_Float16)(a.x + c.x);
        pk.h[1] = (_Float16)(a.y + c.y);
        *(unsigned int*)(Tbuf + ((size_t)b0 * 32 + 1 + t) * EMB + lane * 2) = pk.u;
    }
    {
        const float2 a = *(const float2*)pB0;
        const float2 c = *(const float2*)pB1;
        pk.h[0] = (_Float16)(a.x + c.x);
        pk.h[1] = (_Float16)(a.y + c.y);
        *(unsigned int*)(Tbuf + ((size_t)(b0 + 1) * 32 + 1 + t) * EMB + lane * 2) = pk.u;
    }
}

// ---------------------------------------------------------------------------
// Dot interaction: one wave per sample. T = Tbuf[b] : [32, 64] f16 (rows
// 27-31 garbage but never stored). Z = T T^T via 3 WMMA tiles; strict lower
// triangle (27 rows) -> R[b][64 + i*(i-1)/2 + j]; R[b][0:64] = x; R[b][415]=0.
// ---------------------------------------------------------------------------
__global__ __launch_bounds__(256) void interact_wmma(
    const _Float16* __restrict__ Tbuf, _Float16* __restrict__ R)
{
    const int lane = threadIdx.x & 31;
    const int b = (int)((blockIdx.x * (unsigned)blockDim.x + threadIdx.x) >> 5);
    const _Float16* T = Tbuf + (size_t)b * 32 * EMB;
    _Float16* Rb = R + (size_t)b * 416;

    // copy x (row 0 of T) into R[0:64]
    *(unsigned int*)(Rb + lane * 2) = *(const unsigned int*)(T + lane * 2);
    if (lane == 0) Rb[415] = (_Float16)0.f;

    const int akoff = (lane >= 16) ? 8 : 0;
    const int bkoff = (lane >= 16) ? 16 : 0;
    const int TI[3] = {0, 1, 1};
    const int TJ[3] = {0, 0, 1};

#pragma unroll
    for (int tt = 0; tt < 3; ++tt) {
        const int ti = TI[tt], tj = TJ[tt];
        const int arow = ti * 16 + (lane & 15);   // Z row block
        const int jrow = tj * 16 + (lane & 15);   // Z col block (row of T)
        v8f acc = {0.f, 0.f, 0.f, 0.f, 0.f, 0.f, 0.f, 0.f};
#pragma unroll
        for (int k0 = 0; k0 < EMB; k0 += 32) {
            union { v16h v; v8h h[2]; } a, bb;
            const _Float16* ap = T + (size_t)arow * EMB + k0;
            const _Float16* bp = T + (size_t)jrow * EMB + k0;
            a.h[0]  = *(const v8h*)(ap + akoff);
            a.h[1]  = *(const v8h*)(ap + 16 + akoff);
            bb.h[0] = *(const v8h*)(bp + akoff);
            bb.h[1] = *(const v8h*)(bp + 16 + akoff);
            acc = __builtin_amdgcn_wmma_f32_16x16x32_f16(
                      false, a.v, false, bb.v, (short)0, acc, false, false);
        }
        const int j = tj * 16 + (lane & 15);
        const int ibase = ti * 16 + ((lane >= 16) ? 8 : 0);
#pragma unroll
        for (int r = 0; r < 8; ++r) {
            const int i = ibase + r;
            if (i < 27 && j < i)
                Rb[64 + (i * (i - 1)) / 2 + j] = (_Float16)acc[r];
        }
    }
}

// ---------------------------------------------------------------------------
// Final top layer: out[b] = sigmoid( act[b,:256] . w[:256] + bias ), f32 out.
// One wave per row; shuffle reduction.
// ---------------------------------------------------------------------------
__global__ __launch_bounds__(256) void top_final(
    const _Float16* __restrict__ act, const _Float16* __restrict__ w,
    const float* __restrict__ bias, float* __restrict__ out)
{
    const int lane = threadIdx.x & 31;
    const int row = (int)((blockIdx.x * (unsigned)blockDim.x + threadIdx.x) >> 5);
    const v8h a  = *(const v8h*)(act + (size_t)row * 256 + lane * 8);
    const v8h wv = *(const v8h*)(w + lane * 8);
    float s = 0.f;
#pragma unroll
    for (int i = 0; i < 8; ++i) s += (float)a[i] * (float)wv[i];
#pragma unroll
    for (int off = 16; off > 0; off >>= 1) s += __shfl_down(s, off, 32);
    if (lane == 0) out[row] = 1.f / (1.f + __expf(-(s + bias[0])));
}

// ---------------------------------------------------------------------------
// Workspace layout (offsets in f16 elements, all 32B aligned)
// ---------------------------------------------------------------------------
static const size_t OFF_WB0  = 0;                              // 512*32
static const size_t OFF_WB1  = OFF_WB0 + 512 * 32;             // 256*512
static const size_t OFF_WB2  = OFF_WB1 + 256 * 512;            // 64*256
static const size_t OFF_WT0  = OFF_WB2 + 64 * 256;             // 512*416
static const size_t OFF_WT1  = OFF_WT0 + 512 * 416;            // 256*512
static const size_t OFF_WT2  = OFF_WT1 + 256 * 512;            // 256
static const size_t OFF_XP   = OFF_WT2 + 256;                  // B*32
static const size_t OFF_ACTA = OFF_XP + (size_t)BATCH * 32;    // B*512
static const size_t OFF_ACTB = OFF_ACTA + (size_t)BATCH * 512; // B*256
static const size_t OFF_T    = OFF_ACTB + (size_t)BATCH * 256; // B*32*64
static const size_t OFF_R    = OFF_T + (size_t)BATCH * 32 * 64;// B*416

static inline void launch_cvt(const float* s, _Float16* d, int rows, int sk, int dk,
                              hipStream_t st) {
    long total = (long)rows * dk;
    int blocks = (int)((total + 255) / 256);
    hipLaunchKernelGGL(cvt_pad_f16, dim3(blocks), dim3(256), 0, st, s, d, rows, sk, dk);
}

extern "C" void kernel_launch(void* const* d_in, const int* in_sizes, int n_in,
                              void* d_out, int out_size, void* d_ws, size_t ws_size,
                              hipStream_t stream) {
    const float* dense_x = (const float*)d_in[0];
    const int*   lS_i    = (const int*)d_in[2];
    const float* E       = (const float*)d_in[3];
    const float* bW0 = (const float*)d_in[4];  const float* bb0 = (const float*)d_in[5];
    const float* bW1 = (const float*)d_in[6];  const float* bb1 = (const float*)d_in[7];
    const float* bW2 = (const float*)d_in[8];  const float* bb2 = (const float*)d_in[9];
    const float* tW0 = (const float*)d_in[10]; const float* tb0 = (const float*)d_in[11];
    const float* tW1 = (const float*)d_in[12]; const float* tb1 = (const float*)d_in[13];
    const float* tW2 = (const float*)d_in[14]; const float* tb2 = (const float*)d_in[15];
    float* out = (float*)d_out;
    _Float16* ws = (_Float16*)d_ws;

    _Float16* wb0  = ws + OFF_WB0;
    _Float16* wb1  = ws + OFF_WB1;
    _Float16* wb2  = ws + OFF_WB2;
    _Float16* wt0  = ws + OFF_WT0;
    _Float16* wt1  = ws + OFF_WT1;
    _Float16* wt2  = ws + OFF_WT2;
    _Float16* xp   = ws + OFF_XP;
    _Float16* actA = ws + OFF_ACTA;
    _Float16* actB = ws + OFF_ACTB;
    _Float16* Tb   = ws + OFF_T;
    _Float16* Rb   = ws + OFF_R;

    // --- convert weights / dense input to f16 (with zero K-padding) ---
    launch_cvt(bW0, wb0, 512, 13, 32, stream);
    launch_cvt(bW1, wb1, 256, 512, 512, stream);
    launch_cvt(bW2, wb2, 64, 256, 256, stream);
    launch_cvt(tW0, wt0, 512, 415, 416, stream);
    launch_cvt(tW1, wt1, 256, 512, 512, stream);
    launch_cvt(tW2, wt2, 1, 256, 256, stream);
    launch_cvt(dense_x, xp, BATCH, 13, 32, stream);

    // --- bottom MLP (WMMA, fused bias+ReLU, f16 activations) ---
    hipLaunchKernelGGL(gemm_wmma_relu, dim3(512 / 64, BATCH / 256), dim3(256), 0, stream,
                       xp, 32, wb0, 32, bb0, actA, 512, 32, 1);
    hipLaunchKernelGGL(gemm_wmma_relu, dim3(256 / 64, BATCH / 256), dim3(256), 0, stream,
                       actA, 512, wb1, 512, bb1, actB, 256, 512, 1);
    // layer 2 writes x directly into row 0 of padded T buffer [B,32,64]
    hipLaunchKernelGGL(gemm_wmma_relu, dim3(64 / 64, BATCH / 256), dim3(256), 0, stream,
                       actB, 256, wb2, 256, bb2, Tb, 32 * 64, 256, 1);

    // --- embedding bags into rows 1..26 of T buffer (2 bags per wave) ---
    {
        long waves = (long)NTAB * BATCH / 2;
        int blocks = (int)(waves * 32 / 256);
        hipLaunchKernelGGL(embed_bag, dim3(blocks), dim3(256), 0, stream, E, lS_i, Tb);
    }

    // --- dot interaction -> R [B,416] f16 ---
    hipLaunchKernelGGL(interact_wmma, dim3(BATCH * 32 / 256), dim3(256), 0, stream, Tb, Rb);

    // --- top MLP ---
    hipLaunchKernelGGL(gemm_wmma_relu, dim3(512 / 64, BATCH / 256), dim3(256), 0, stream,
                       Rb, 416, wt0, 416, tb0, actA, 512, 416, 1);
    hipLaunchKernelGGL(gemm_wmma_relu, dim3(256 / 64, BATCH / 256), dim3(256), 0, stream,
                       actA, 512, wt1, 512, tb1, actB, 256, 512, 1);
    hipLaunchKernelGGL(top_final, dim3(BATCH * 32 / 256), dim3(256), 0, stream,
                       actB, wt2, tb2, out);

    (void)in_sizes; (void)n_in; (void)out_size; (void)ws_size;
}
